// MSA_30064771072887
// MI455X (gfx1250) — matrix operand order repaired and checked by reading, not compiled, not run on hardware
//
#include <hip/hip_runtime.h>
#include <math.h>

#define EMBED 768
#define HEADS 12
#define DHEAD 64
#define SEQ   2048
#define BATCH 2
#define M_TOT (BATCH*SEQ)   // 4096
#define QKV_N (3*EMBED)     // 2304

typedef __attribute__((ext_vector_type(16))) _Float16 v16h;
typedef __attribute__((ext_vector_type(8)))  _Float16 v8h;
typedef __attribute__((ext_vector_type(4)))  _Float16 v4h;
typedef __attribute__((ext_vector_type(8)))  float    v8f;

// ---- WMMA fragment helpers (wave32 CDNA5 layouts, see ISA 7.12.2) ----

// A-matrix 16x32 f16: lane holds row (lane&15); elems 0..7 = K[h0..h0+7],
// elems 8..15 = K[h0+16..h0+23], h0 = (lane<16)?0:8.
__device__ __forceinline__ v16h frag_a(const _Float16* p, int h0) {
  v8h lo = *(const v8h*)(p + h0);
  v8h hi = *(const v8h*)(p + 16 + h0);
  v16h r;
#pragma unroll
  for (int i = 0; i < 8; ++i) { r[i] = lo[i]; r[i + 8] = hi[i]; }
  return r;
}

// B-matrix 32x16 f16: lane holds column (lane&15); elems 0..7 = K[boff/2..boff/2+7],
// elems 8..15 = K[16+boff/2..16+boff/2+7], boff = (lane<16)?0:16.  p = N-major (K-contiguous) row for this column.
__device__ __forceinline__ v16h frag_b(const _Float16* p, int boff) {
  v8h lo = *(const v8h*)(p + (boff >> 1));
  v8h hi = *(const v8h*)(p + 16 + (boff >> 1));
  v16h r;
#pragma unroll
  for (int i = 0; i < 8; ++i) { r[i] = lo[i]; r[i + 8] = hi[i]; }
  return r;
}

__device__ __forceinline__ v8f wmma_f16(v16h a, v16h b, v8f c) {
  return __builtin_amdgcn_wmma_f32_16x16x32_f16(
      /*neg_a=*/false, a, /*neg_b=*/false, b,
      /*c_mod=*/(short)0, c, /*reuse_a=*/false, /*reuse_b=*/false);
}

// ---- f32 -> f16 conversion (4 elements / thread) ----
__global__ void cvt_kernel(const float* __restrict__ in,
                           _Float16* __restrict__ out, int n4) {
  int i = blockIdx.x * blockDim.x + threadIdx.x;
  if (i < n4) {
    float4 v = ((const float4*)in)[i];
    v4h h = {(_Float16)v.x, (_Float16)v.y, (_Float16)v.z, (_Float16)v.w};
    ((v4h*)out)[i] = h;
  }
}

// ---- QKV projection: qkv = x @ w_qkv^T + b ; scatter into Q,K (b,h,n,d) and
// ---- Vt (b,h,d,n).  Q pre-scaled by 1/sqrt(EMBED).  Wave tile: 32x64. ----
__global__ void qkv_gemm(const _Float16* __restrict__ xh,
                         const _Float16* __restrict__ wh,
                         const float* __restrict__ bias,
                         _Float16* __restrict__ Qb,
                         _Float16* __restrict__ Kb,
                         _Float16* __restrict__ Vt) {
  const int lane = threadIdx.x & 31;
  const int wid  = (blockIdx.x * blockDim.x + threadIdx.x) >> 5;
  const int NT   = QKV_N / 64;                       // 36 col tiles
  if (wid >= (M_TOT / 32) * NT) return;
  const int mt = wid / NT, nt = wid % NT;
  const int m0 = mt * 32, n0 = nt * 64;
  const int cl = lane & 15;
  const int h0 = (lane < 16) ? 0 : 8;
  const int bo = (lane < 16) ? 0 : 16;

  v8f acc[2][4] = {};
  const _Float16* arow0 = xh + (m0 + cl) * EMBED;
  const _Float16* arow1 = arow0 + 16 * EMBED;
  const _Float16* brow[4];
#pragma unroll
  for (int j = 0; j < 4; ++j) brow[j] = wh + (n0 + j * 16 + cl) * EMBED;

  for (int k0 = 0; k0 < EMBED; k0 += 32) {
    v16h a0 = frag_a(arow0 + k0, h0);
    v16h a1 = frag_a(arow1 + k0, h0);
#pragma unroll
    for (int j = 0; j < 4; ++j) {
      v16h b = frag_b(brow[j] + k0, bo);
      acc[0][j] = wmma_f16(a0, b, acc[0][j]);
      acc[1][j] = wmma_f16(a1, b, acc[1][j]);
    }
  }

  const int rb = (lane < 16) ? 0 : 8;
  const float scale = 0.03608439182435161f;          // 1/sqrt(768)
#pragma unroll
  for (int mi = 0; mi < 2; ++mi)
#pragma unroll
    for (int j = 0; j < 4; ++j) {
      int col = n0 + j * 16 + cl;
      int h = col / 192, rem = col % 192;
      int s = rem / 64, dd = rem % 64;
      float bv = bias[col];
#pragma unroll
      for (int r = 0; r < 8; ++r) {
        int gm = m0 + mi * 16 + rb + r;
        int b = gm >> 11, n = gm & (SEQ - 1);
        float val = acc[mi][j][r] + bv;
        if (s == 0)
          Qb[((b * HEADS + h) * SEQ + n) * DHEAD + dd] = (_Float16)(val * scale);
        else if (s == 1)
          Kb[((b * HEADS + h) * SEQ + n) * DHEAD + dd] = (_Float16)val;
        else
          Vt[((b * HEADS + h) * DHEAD + dd) * SEQ + n] = (_Float16)val;
      }
    }
}

// ---- Flash attention: one wave = 16 query rows of one (b,h). ----
__global__ void attn_kernel(const _Float16* __restrict__ Qb,
                            const _Float16* __restrict__ Kb,
                            const _Float16* __restrict__ Vt,
                            _Float16* __restrict__ ctx) {
  const int lane  = threadIdx.x & 31;
  const int wslot = threadIdx.x >> 5;                // 0..7 (wave in block)
  const int wid   = (blockIdx.x * blockDim.x + threadIdx.x) >> 5;
  if (wid >= BATCH * HEADS * (SEQ / 16)) return;
  const int qt = wid & 127;                          // q tile within head
  const int bh = wid >> 7;                           // 0..23
  const int q0 = qt * 16;

  const _Float16* Qp = Qb + (size_t)bh * SEQ * DHEAD;
  const _Float16* Kp = Kb + (size_t)bh * SEQ * DHEAD;
  const _Float16* Vp = Vt + (size_t)bh * DHEAD * SEQ;

  __shared__ __align__(16) float    Sbuf[8][16 * 32];
  __shared__ __align__(16) _Float16 Pbuf[8][16 * 32];
  __shared__ __align__(16) float    Abuf[8][16];
  float*    S  = Sbuf[wslot];
  _Float16* P  = Pbuf[wslot];
  float*    AL = Abuf[wslot];

  const int cl = lane & 15;
  const int h0 = (lane < 16) ? 0 : 8;
  const int bo = (lane < 16) ? 0 : 16;
  const int rb = (lane < 16) ? 0 : 8;

  // Q A-frags (d = 64 -> two K-steps of 32), loaded once.
  const _Float16* qrow = Qp + (q0 + cl) * DHEAD;
  v16h qa0 = frag_a(qrow, h0);
  v16h qa1 = frag_a(qrow + 32, h0);

  v8f o[4] = {};                                     // ctx accum 16x64 f32
  float m = -1e30f, l = 0.f;                         // row stats (lanes 0..15)

  for (int kv0 = 0; kv0 < SEQ; kv0 += 32) {
    // S(16x32) = Q * K^T  (K rows are K-contiguous -> B-frags)
    const _Float16* krow0 = Kp + (kv0 + cl) * DHEAD;
    const _Float16* krow1 = krow0 + 16 * DHEAD;
    v8f s0 = {}, s1 = {};
    s0 = wmma_f16(qa0, frag_b(krow0, bo), s0);
    s0 = wmma_f16(qa1, frag_b(krow0 + 32, bo), s0);
    s1 = wmma_f16(qa0, frag_b(krow1, bo), s1);
    s1 = wmma_f16(qa1, frag_b(krow1 + 32, bo), s1);

#pragma unroll
    for (int r = 0; r < 8; ++r) {
      S[(rb + r) * 32 + cl]      = s0[r];
      S[(rb + r) * 32 + 16 + cl] = s1[r];
    }
    __builtin_amdgcn_wave_barrier();

    if (lane < 16) {                                 // one row per lane
      const float* srow = S + lane * 32;
      float sv[32];
#pragma unroll
      for (int i = 0; i < 8; ++i) {
        float4 t = ((const float4*)srow)[i];
        sv[4 * i] = t.x; sv[4 * i + 1] = t.y; sv[4 * i + 2] = t.z; sv[4 * i + 3] = t.w;
      }
      float bm = sv[0];
#pragma unroll
      for (int i = 1; i < 32; ++i) bm = fmaxf(bm, sv[i]);
      float nm    = fmaxf(m, bm);
      float alpha = __expf(m - nm);
      float ssum  = 0.f;
      _Float16* prow = P + lane * 32;
#pragma unroll
      for (int i = 0; i < 4; ++i) {
        v8h ph;
#pragma unroll
        for (int j = 0; j < 8; ++j) {
          float pv = __expf(sv[8 * i + j] - nm);
          ssum += pv;
          ph[j] = (_Float16)pv;
        }
        ((v8h*)prow)[i] = ph;
      }
      l = l * alpha + ssum;
      m = nm;
      AL[lane] = alpha;
    }
    __builtin_amdgcn_wave_barrier();

    // rescale accumulator by alpha of each row
    float4 a0 = ((const float4*)(AL + rb))[0];
    float4 a1 = ((const float4*)(AL + rb))[1];
    float af[8] = {a0.x, a0.y, a0.z, a0.w, a1.x, a1.y, a1.z, a1.w};
#pragma unroll
    for (int t = 0; t < 4; ++t)
#pragma unroll
      for (int r = 0; r < 8; ++r) o[t][r] *= af[r];

    // ctx += P(16x32) * V(32x64); Vt rows are kv-contiguous -> B-frags
    v16h pa = frag_a(P + cl * 32, h0);
#pragma unroll
    for (int t = 0; t < 4; ++t) {
      v16h vb = frag_b(Vp + (t * 16 + cl) * SEQ + kv0, bo);
      o[t] = wmma_f16(pa, vb, o[t]);
    }
    __builtin_amdgcn_wave_barrier();
  }

  // finalize: divide by l, write ctx[b, n, h*64+dd] as f16
  if (lane < 16) AL[lane] = l;
  __builtin_amdgcn_wave_barrier();
  float4 l0 = ((const float4*)(AL + rb))[0];
  float4 l1 = ((const float4*)(AL + rb))[1];
  float lv[8] = {1.f / l0.x, 1.f / l0.y, 1.f / l0.z, 1.f / l0.w,
                 1.f / l1.x, 1.f / l1.y, 1.f / l1.z, 1.f / l1.w};
  const int bb = bh / HEADS, hh = bh % HEADS;
#pragma unroll
  for (int t = 0; t < 4; ++t) {
    int col = hh * DHEAD + t * 16 + cl;
#pragma unroll
    for (int r = 0; r < 8; ++r) {
      int grow = bb * SEQ + q0 + rb + r;
      ctx[(size_t)grow * EMBED + col] = (_Float16)(o[t][r] * lv[r]);
    }
  }
}

// ---- Output projection: out = ctx @ w_o^T + b_o (f32 out). Wave tile 32x64. ----
__global__ void out_gemm(const _Float16* __restrict__ ch,
                         const _Float16* __restrict__ wh,
                         const float* __restrict__ bias,
                         float* __restrict__ out) {
  const int lane = threadIdx.x & 31;
  const int wid  = (blockIdx.x * blockDim.x + threadIdx.x) >> 5;
  const int NT   = EMBED / 64;                       // 12
  if (wid >= (M_TOT / 32) * NT) return;
  const int mt = wid / NT, nt = wid % NT;
  const int m0 = mt * 32, n0 = nt * 64;
  const int cl = lane & 15;
  const int h0 = (lane < 16) ? 0 : 8;
  const int bo = (lane < 16) ? 0 : 16;

  v8f acc[2][4] = {};
  const _Float16* arow0 = ch + (m0 + cl) * EMBED;
  const _Float16* arow1 = arow0 + 16 * EMBED;
  const _Float16* brow[4];
#pragma unroll
  for (int j = 0; j < 4; ++j) brow[j] = wh + (n0 + j * 16 + cl) * EMBED;

  for (int k0 = 0; k0 < EMBED; k0 += 32) {
    v16h a0 = frag_a(arow0 + k0, h0);
    v16h a1 = frag_a(arow1 + k0, h0);
#pragma unroll
    for (int j = 0; j < 4; ++j) {
      v16h b = frag_b(brow[j] + k0, bo);
      acc[0][j] = wmma_f16(a0, b, acc[0][j]);
      acc[1][j] = wmma_f16(a1, b, acc[1][j]);
    }
  }

  const int rb = (lane < 16) ? 0 : 8;
#pragma unroll
  for (int mi = 0; mi < 2; ++mi)
#pragma unroll
    for (int j = 0; j < 4; ++j) {
      int col = n0 + j * 16 + cl;
      float bv = bias[col];
#pragma unroll
      for (int r = 0; r < 8; ++r) {
        int gm = m0 + mi * 16 + rb + r;
        out[(size_t)gm * EMBED + col] = acc[mi][j][r] + bv;
      }
    }
}

extern "C" void kernel_launch(void* const* d_in, const int* in_sizes, int n_in,
                              void* d_out, int out_size, void* d_ws, size_t ws_size,
                              hipStream_t stream) {
  const float* x    = (const float*)d_in[0];
  const float* wqkv = (const float*)d_in[1];
  const float* bqkv = (const float*)d_in[2];
  const float* wo   = (const float*)d_in[3];
  const float* bo   = (const float*)d_in[4];
  float* out = (float*)d_out;

  char* p = (char*)d_ws;
  _Float16* xh   = (_Float16*)p; p += (size_t)M_TOT * EMBED * 2;
  _Float16* wqh  = (_Float16*)p; p += (size_t)QKV_N * EMBED * 2;
  _Float16* woh  = (_Float16*)p; p += (size_t)EMBED * EMBED * 2;
  _Float16* Qb   = (_Float16*)p; p += (size_t)M_TOT * EMBED * 2;
  _Float16* Kb   = (_Float16*)p; p += (size_t)M_TOT * EMBED * 2;
  _Float16* Vt   = (_Float16*)p; p += (size_t)M_TOT * EMBED * 2;
  _Float16* ctxh = (_Float16*)p; p += (size_t)M_TOT * EMBED * 2;

  int n4 = M_TOT * EMBED / 4;
  cvt_kernel<<<(n4 + 255) / 256, 256, 0, stream>>>(x, xh, n4);
  n4 = QKV_N * EMBED / 4;
  cvt_kernel<<<(n4 + 255) / 256, 256, 0, stream>>>(wqkv, wqh, n4);
  n4 = EMBED * EMBED / 4;
  cvt_kernel<<<(n4 + 255) / 256, 256, 0, stream>>>(wo, woh, n4);

  // waves: (4096/32)*(2304/64) = 4608 -> 576 blocks of 8 waves
  qkv_gemm<<<576, 256, 0, stream>>>(xh, wqh, bqkv, Qb, Kb, Vt);
  // waves: 2*12*(2048/16) = 3072 -> 384 blocks of 8 waves
  attn_kernel<<<384, 256, 0, stream>>>(Qb, Kb, Vt, ctxh);
  // waves: (4096/32)*(768/64) = 1536 -> 192 blocks of 8 waves
  out_gemm<<<192, 256, 0, stream>>>(ctxh, woh, bo, out);
}
